// GIN_40475771797957
// MI455X (gfx1250) — matrix operand order repaired and checked by reading, not compiled, not run on hardware
//
#include <hip/hip_runtime.h>
#include <hip/hip_bf16.h>

// ---------------- problem constants (match reference) ----------------
#define N_NODES   50000
#define N_EDGES   800000
#define EMB       128
#define HID       256
#define N_LAYERS  5
#define NUM_GRAPHS 128
#define NUM_TASK  2
#define BN_EPS    1e-5f

typedef __attribute__((ext_vector_type(16))) __bf16 v16bf;
typedef __attribute__((ext_vector_type(8)))  __bf16 v8bf;
typedef __attribute__((ext_vector_type(4)))  __bf16 v4bf;
typedef __attribute__((ext_vector_type(8)))  float  v8f;

#define CEILDIV(a, b) (((a) + (b) - 1) / (b))

// Native fp32 global atomic add (no return -> STOREcnt path, no CAS loop)
__device__ __forceinline__ void atomAdd(float* p, float v) {
    __hip_atomic_fetch_add(p, v, __ATOMIC_RELAXED, __HIP_MEMORY_SCOPE_AGENT);
}

// ---------------- utility kernels ----------------
__global__ void zero_f32(float* p, long n) {
    long i = (long)blockIdx.x * blockDim.x + threadIdx.x;
    if (i < n) p[i] = 0.0f;
}

// Convert weights [L][K][C] f32 -> transposed bf16 [L][C][K] (B-fragment friendly)
__global__ void conv_wt_all(const float* __restrict__ W, __bf16* __restrict__ Wt,
                            int L, int K, int C) {
    long i = (long)blockIdx.x * blockDim.x + threadIdx.x;
    long tot = (long)L * K * C;
    if (i >= tot) return;
    int per = K * C;
    int l = (int)(i / per);
    int rem = (int)(i % per);
    int k = rem / C;
    int c = rem % C;
    Wt[(long)l * per + (long)c * K + k] = (__bf16)W[i];
}

// ---------------- embedding / message passing (4 channels/thread, b128) -------
__global__ void node_embed(const int* __restrict__ node_attr,
                           const float* __restrict__ atom_emb,
                           float* __restrict__ h) {
    long i = (long)blockIdx.x * blockDim.x + threadIdx.x;  // N_NODES * 32
    if (i >= (long)N_NODES * 32) return;
    int n = (int)(i >> 5);
    int t = (int)((i & 31) << 2);
    const int* attr = node_attr + n * 9;
    float4 s = make_float4(0.f, 0.f, 0.f, 0.f);
#pragma unroll
    for (int f = 0; f < 9; ++f) {
        const float4 v = *(const float4*)(atom_emb + (f * 64 + attr[f]) * EMB + t);
        s.x += v.x; s.y += v.y; s.z += v.z; s.w += v.w;
    }
    *(float4*)(h + (long)n * EMB + t) = s;
}

__global__ void rst_scale(const float* __restrict__ h, const float* __restrict__ eps,
                          int layer, float* __restrict__ rst) {
    long i = (long)blockIdx.x * blockDim.x + threadIdx.x;  // (N*EMB)/4
    if (i >= (long)N_NODES * EMB / 4) return;
    const float s = 1.0f + eps[layer];
    const float4 v = *(const float4*)(h + i * 4);
    float4 r;
    r.x = s * v.x; r.y = s * v.y; r.z = s * v.z; r.w = s * v.w;
    *(float4*)(rst + i * 4) = r;
}

// msg = h[src] + edge_embed(edge_attr); rst[dst] += msg   (bond table is cache-resident)
__global__ void scatter_edges(const int* __restrict__ src, const int* __restrict__ dst,
                              const int* __restrict__ edge_attr,
                              const float* __restrict__ bond_emb,
                              const float* __restrict__ h, float* __restrict__ rst) {
    long i = (long)blockIdx.x * blockDim.x + threadIdx.x;  // N_EDGES * 32
    if (i >= (long)N_EDGES * 32) return;
    int e = (int)(i >> 5);
    int t = (int)((i & 31) << 2);
    const int* ea = edge_attr + e * 3;
    float4 m = *(const float4*)(h + (long)src[e] * EMB + t);
#pragma unroll
    for (int f = 0; f < 3; ++f) {
        const float4 bv = *(const float4*)(bond_emb + (f * 8 + ea[f]) * EMB + t);
        m.x += bv.x; m.y += bv.y; m.z += bv.z; m.w += bv.w;
    }
    float* o = rst + (long)dst[e] * EMB + t;
    atomAdd(o + 0, m.x);
    atomAdd(o + 1, m.y);
    atomAdd(o + 2, m.z);
    atomAdd(o + 3, m.w);
}

// ---------------- WMMA GEMM: C[M][NC] = A[M][K] * B[K][NC] + bias ----------------
// Bt is bf16, pre-transposed to [NC][K]. A is f32 (converted on load) or bf16.
// Each wave computes a 16x64 strip: 4 wmma tiles, A fragment reused 4x per K-step.
template <bool A_IS_F32, int K, int NC>
__global__ __launch_bounds__(256) void gemm_bias_wmma(
    const void* __restrict__ Aptr, const __bf16* __restrict__ Bt,
    const float* __restrict__ bias, float* __restrict__ Cout, int Mrows) {
    constexpr int CG = NC / 64;                 // column groups of 64
    const int lane = threadIdx.x & 31;
    const int wave = blockIdx.x * 8 + (threadIdx.x >> 5);
    const int MT = Mrows >> 4;                  // 16-row tiles (M divisible by 16)
    if (wave >= MT * CG) return;
    const int row0 = (wave / CG) * 16;
    const int col0 = (wave % CG) * 64;
    const int m    = lane & 15;
    const int half = lane >> 4;

    v8f acc[4] = {};

    const int arow = (row0 + m) * K;

    for (int ks = 0; ks < K; ks += 32) {
        // ---- A fragment: 16x32 bf16; lane m, K = {half*8..+7, 16+half*8..+7}
        v16bf a;
        if constexpr (A_IS_F32) {
            const float* A = (const float*)Aptr;
            const float4* p0 = (const float4*)(A + arow + ks + half * 8);
            const float4* p1 = (const float4*)(A + arow + ks + 16 + half * 8);
            float4 x0 = p0[0], x1 = p0[1];
            float4 x2 = p1[0], x3 = p1[1];
            a[0]  = (__bf16)x0.x; a[1]  = (__bf16)x0.y; a[2]  = (__bf16)x0.z; a[3]  = (__bf16)x0.w;
            a[4]  = (__bf16)x1.x; a[5]  = (__bf16)x1.y; a[6]  = (__bf16)x1.z; a[7]  = (__bf16)x1.w;
            a[8]  = (__bf16)x2.x; a[9]  = (__bf16)x2.y; a[10] = (__bf16)x2.z; a[11] = (__bf16)x2.w;
            a[12] = (__bf16)x3.x; a[13] = (__bf16)x3.y; a[14] = (__bf16)x3.z; a[15] = (__bf16)x3.w;
        } else {
            const __bf16* A = (const __bf16*)Aptr;
            const v8bf lo = *(const v8bf*)(A + arow + ks + half * 8);
            const v8bf hi = *(const v8bf*)(A + arow + ks + 16 + half * 8);
#pragma unroll
            for (int i = 0; i < 8; ++i) { a[i] = lo[i]; a[8 + i] = hi[i]; }
        }
        // ---- 4 B fragments + WMMAs (B: lane n, K = {half*8..+7, 16+half*8..+7})
#pragma unroll
        for (int nt = 0; nt < 4; ++nt) {
            const __bf16* bptr = Bt + (col0 + nt * 16 + m) * K + ks + half * 8;
            const v8bf b0 = *(const v8bf*)(bptr);
            const v8bf b1 = *(const v8bf*)(bptr + 16);
            v16bf b;
#pragma unroll
            for (int i = 0; i < 8; ++i) { b[i] = b0[i]; b[8 + i] = b1[i]; }
            acc[nt] = __builtin_amdgcn_wmma_f32_16x16x32_bf16(
                false, a, false, b, (short)0, acc[nt], false, false);
        }
    }

    // ---- epilogue: D layout row = row0 + 8*half + r, col = col0 + nt*16 + m
#pragma unroll
    for (int nt = 0; nt < 4; ++nt) {
        const int col = col0 + nt * 16 + m;
        const float bv = bias[col];
#pragma unroll
        for (int r = 0; r < 8; ++r) {
            Cout[(row0 + half * 8 + r) * NC + col] = acc[nt][r] + bv;
        }
    }
}

// ---------------- batchnorm (training mode, biased var) ----------------
// column sums & sum-of-squares; blockDim.x == C/4, each block covers 128 rows
__global__ void colstats(const float* __restrict__ Z, float* __restrict__ sum,
                         float* __restrict__ sumsq, int nrows) {
    const int C = blockDim.x * 4;
    const int c = threadIdx.x * 4;
    int r0 = blockIdx.x * 128;
    int rend = r0 + 128;
    if (rend > nrows) rend = nrows;
    float4 s = make_float4(0.f, 0.f, 0.f, 0.f);
    float4 s2 = make_float4(0.f, 0.f, 0.f, 0.f);
    for (int r = r0; r < rend; ++r) {
        const float4 v = *(const float4*)(Z + (long)r * C + c);
        s.x += v.x; s.y += v.y; s.z += v.z; s.w += v.w;
        s2.x += v.x * v.x; s2.y += v.y * v.y; s2.z += v.z * v.z; s2.w += v.w * v.w;
    }
    atomAdd(&sum[c + 0], s.x); atomAdd(&sum[c + 1], s.y);
    atomAdd(&sum[c + 2], s.z); atomAdd(&sum[c + 3], s.w);
    atomAdd(&sumsq[c + 0], s2.x); atomAdd(&sumsq[c + 1], s2.y);
    atomAdd(&sumsq[c + 2], s2.z); atomAdd(&sumsq[c + 3], s2.w);
}

__global__ void bn_finalize(const float* __restrict__ sum, const float* __restrict__ sumsq,
                            const float* __restrict__ gamma, const float* __restrict__ beta,
                            float* __restrict__ scale, float* __restrict__ shift,
                            float invN) {
    int c = threadIdx.x;
    float mu = sum[c] * invN;
    float var = sumsq[c] * invN - mu * mu;
    float a = gamma[c] * rsqrtf(var + BN_EPS);
    scale[c] = a;
    shift[c] = beta[c] - mu * a;
}

// y = relu(scale*x + shift), f32 out (4/thread)
__global__ void bn_relu_f32(const float* __restrict__ Z, const float* __restrict__ scale,
                            const float* __restrict__ shift, float* __restrict__ out,
                            long total4, int cmask) {
    long i = (long)blockIdx.x * blockDim.x + threadIdx.x;
    if (i >= total4) return;
    long base = i * 4;
    int c = (int)(base & cmask);
    const float4 z = *(const float4*)(Z + base);
    const float4 sc = *(const float4*)(scale + c);
    const float4 sh = *(const float4*)(shift + c);
    float4 r;
    r.x = fmaxf(fmaf(z.x, sc.x, sh.x), 0.f);
    r.y = fmaxf(fmaf(z.y, sc.y, sh.y), 0.f);
    r.z = fmaxf(fmaf(z.z, sc.z, sh.z), 0.f);
    r.w = fmaxf(fmaf(z.w, sc.w, sh.w), 0.f);
    *(float4*)(out + base) = r;
}

// y = relu(scale*x + shift), bf16 out (4/thread, packed 8B store)
__global__ void bn_relu_bf16(const float* __restrict__ Z, const float* __restrict__ scale,
                             const float* __restrict__ shift, __bf16* __restrict__ out,
                             long total4, int cmask) {
    long i = (long)blockIdx.x * blockDim.x + threadIdx.x;
    if (i >= total4) return;
    long base = i * 4;
    int c = (int)(base & cmask);
    const float4 z = *(const float4*)(Z + base);
    const float4 sc = *(const float4*)(scale + c);
    const float4 sh = *(const float4*)(shift + c);
    v4bf r;
    r[0] = (__bf16)fmaxf(fmaf(z.x, sc.x, sh.x), 0.f);
    r[1] = (__bf16)fmaxf(fmaf(z.y, sc.y, sh.y), 0.f);
    r[2] = (__bf16)fmaxf(fmaf(z.z, sc.z, sh.z), 0.f);
    r[3] = (__bf16)fmaxf(fmaf(z.w, sc.w, sh.w), 0.f);
    *(v4bf*)(out + base) = r;
}

// ---------------- readout ----------------
__global__ void graph_scatter(const float* __restrict__ h, const int* __restrict__ node2graph,
                              float* __restrict__ gsum, float* __restrict__ gcnt) {
    long i = (long)blockIdx.x * blockDim.x + threadIdx.x;  // N_NODES * 32
    if (i >= (long)N_NODES * 32) return;
    int n = (int)(i >> 5);
    int t = (int)((i & 31) << 2);
    int g = node2graph[n];
    const float4 v = *(const float4*)(h + (long)n * EMB + t);
    float* o = gsum + g * EMB + t;
    atomAdd(o + 0, v.x);
    atomAdd(o + 1, v.y);
    atomAdd(o + 2, v.z);
    atomAdd(o + 3, v.w);
    if ((i & 31) == 0) atomAdd(&gcnt[g], 1.0f);
}

__global__ void readout(const float* __restrict__ gsum, const float* __restrict__ gcnt,
                        const float* __restrict__ Wp, const float* __restrict__ bp,
                        float* __restrict__ out) {
    int i = threadIdx.x;  // NUM_GRAPHS * NUM_TASK == 256
    if (i >= NUM_GRAPHS * NUM_TASK) return;
    int g = i >> 1;
    int j = i & 1;
    float cnt = fmaxf(gcnt[g], 1.0f);
    float s = 0.0f;
    for (int c = 0; c < EMB; ++c)
        s += (gsum[g * EMB + c] / cnt) * Wp[c * NUM_TASK + j];
    out[i] = s + bp[j];
}

// ---------------- driver ----------------
extern "C" void kernel_launch(void* const* d_in, const int* in_sizes, int n_in,
                              void* d_out, int out_size, void* d_ws, size_t ws_size,
                              hipStream_t stream) {
    const int*   node_attr  = (const int*)  d_in[0];
    const int*   edge_attr  = (const int*)  d_in[1];
    const int*   src        = (const int*)  d_in[2];
    const int*   dst        = (const int*)  d_in[3];
    const int*   node2graph = (const int*)  d_in[4];
    const float* atom_emb   = (const float*)d_in[5];
    const float* bond_emb   = (const float*)d_in[6];
    const float* eps        = (const float*)d_in[7];
    const float* W1         = (const float*)d_in[8];
    const float* b1         = (const float*)d_in[9];
    const float* g1         = (const float*)d_in[10];
    const float* be1        = (const float*)d_in[11];
    const float* W2         = (const float*)d_in[12];
    const float* b2         = (const float*)d_in[13];
    const float* g2         = (const float*)d_in[14];
    const float* be2        = (const float*)d_in[15];
    const float* Wp         = (const float*)d_in[16];
    const float* bp         = (const float*)d_in[17];
    float* out = (float*)d_out;

    // -------- workspace layout --------
    float*  h     = (float*)d_ws;                                   // N * 128 f32
    float*  rst   = h + (size_t)N_NODES * EMB;                      // N * 128 f32 (reused as z2)
    float*  z1    = rst + (size_t)N_NODES * EMB;                    // N * 256 f32
    __bf16* z1bf  = (__bf16*)(z1 + (size_t)N_NODES * HID);          // N * 256 bf16
    float*  stats = (float*)(z1bf + (size_t)N_NODES * HID);         // 1024 f32
    float*  s_sum   = stats;
    float*  s_sumsq = stats + 256;
    float*  s_scale = stats + 512;
    float*  s_shift = stats + 768;
    __bf16* w1t   = (__bf16*)(stats + 1024);                        // 5*128*256 bf16 [C][K]
    __bf16* w2t   = w1t + (size_t)N_LAYERS * EMB * HID;             // 5*256*128 bf16 [C][K]
    float*  gsum  = (float*)(w2t + (size_t)N_LAYERS * HID * EMB);   // 128*128 f32
    float*  gcnt  = gsum + NUM_GRAPHS * EMB;                        // 128 f32

    const long nodeV4 = (long)N_NODES * 32;        // node elems / 4
    const long hidV4  = (long)N_NODES * HID / 4;
    const long edgeV4 = (long)N_EDGES * 32;        // edge elems / 4

    // -------- weight convert + transpose (all layers, once per call) --------
    {
        long tot1 = (long)N_LAYERS * EMB * HID;
        conv_wt_all<<<CEILDIV(tot1, 256), 256, 0, stream>>>(W1, w1t, N_LAYERS, EMB, HID);
        long tot2 = (long)N_LAYERS * HID * EMB;
        conv_wt_all<<<CEILDIV(tot2, 256), 256, 0, stream>>>(W2, w2t, N_LAYERS, HID, EMB);
    }

    // -------- encoder --------
    node_embed<<<CEILDIV(nodeV4, 256), 256, 0, stream>>>(node_attr, atom_emb, h);

    const int gemm1_blocks = CEILDIV((N_NODES / 16) * (HID / 64), 8);  // 12500 waves
    const int gemm2_blocks = CEILDIV((N_NODES / 16) * (EMB / 64), 8);  //  6250 waves
    const int statBlocks = CEILDIV(N_NODES, 128);

    // -------- GIN layers --------
    for (int l = 0; l < N_LAYERS; ++l) {
        // rst = (1+eps)*h; then scatter-add messages
        rst_scale<<<CEILDIV(nodeV4, 256), 256, 0, stream>>>(h, eps, l, rst);
        scatter_edges<<<CEILDIV(edgeV4, 256), 256, 0, stream>>>(
            src, dst, edge_attr, bond_emb, h, rst);

        // z1 = rst @ W1 + b1   (bf16 WMMA, A converted on load)
        gemm_bias_wmma<true, EMB, HID><<<gemm1_blocks, 256, 0, stream>>>(
            rst, w1t + (size_t)l * EMB * HID, b1 + l * HID, z1, N_NODES);

        // BN1 + ReLU -> bf16 (GEMM2 input)
        zero_f32<<<2, 256, 0, stream>>>(stats, 512);
        colstats<<<statBlocks, HID / 4, 0, stream>>>(z1, s_sum, s_sumsq, N_NODES);
        bn_finalize<<<1, HID, 0, stream>>>(s_sum, s_sumsq, g1 + l * HID, be1 + l * HID,
                                           s_scale, s_shift, 1.0f / N_NODES);
        bn_relu_bf16<<<CEILDIV(hidV4, 256), 256, 0, stream>>>(
            z1, s_scale, s_shift, z1bf, hidV4, HID - 1);

        // z2 = z1bf @ W2 + b2  (into rst buffer)
        gemm_bias_wmma<false, HID, EMB><<<gemm2_blocks, 256, 0, stream>>>(
            z1bf, w2t + (size_t)l * HID * EMB, b2 + l * EMB, rst, N_NODES);

        // BN2 + ReLU -> h (f32)
        zero_f32<<<2, 256, 0, stream>>>(stats, 512);
        colstats<<<statBlocks, EMB / 4, 0, stream>>>(rst, s_sum, s_sumsq, N_NODES);
        bn_finalize<<<1, EMB, 0, stream>>>(s_sum, s_sumsq, g2 + l * EMB, be2 + l * EMB,
                                           s_scale, s_shift, 1.0f / N_NODES);
        bn_relu_f32<<<CEILDIV(nodeV4, 256), 256, 0, stream>>>(
            rst, s_scale, s_shift, h, nodeV4, EMB - 1);
    }

    // -------- readout --------
    zero_f32<<<CEILDIV(NUM_GRAPHS * EMB + NUM_GRAPHS, 256), 256, 0, stream>>>(
        gsum, NUM_GRAPHS * EMB + NUM_GRAPHS);
    graph_scatter<<<CEILDIV(nodeV4, 256), 256, 0, stream>>>(h, node2graph, gsum, gcnt);
    readout<<<1, 256, 0, stream>>>(gsum, gcnt, Wp, bp, out);
}
